// SwinTransformerBlock_65146063946573
// MI455X (gfx1250) — matrix operand order repaired and checked by reading, not compiled, not run on hardware
//
#include <hip/hip_runtime.h>

// ---------- types ----------
typedef __bf16 bf16t;
typedef bf16t v16bf __attribute__((ext_vector_type(16)));
typedef float v8f   __attribute__((ext_vector_type(8)));

struct Q4 { unsigned int x, y, z, w; };
struct Q8 { Q4 lo, hi; };

__device__ __forceinline__ v16bf load_a_tile(const unsigned short* p0) {
  // A-matrix 16x32 bf16 per-lane: elements 0..7 = K base..base+7, 8..15 = K base+16..base+23
  Q8 s; s.lo = *(const Q4*)p0; s.hi = *(const Q4*)(p0 + 16);
  return __builtin_bit_cast(v16bf, s);
}
__device__ __forceinline__ v16bf load_b_tile(const unsigned short* p) {
  // B-matrix 32x16 bf16 per-lane: elements 0..7 = K base..base+7, 8..15 = K base+16..base+23
  Q8 s; s.lo = *(const Q4*)p; s.hi = *(const Q4*)(p + 16);
  return __builtin_bit_cast(v16bf, s);
}
__device__ __forceinline__ v8f vzero8() {
  v8f z = {0.f,0.f,0.f,0.f,0.f,0.f,0.f,0.f}; return z;
}
__device__ __forceinline__ v8f wmma_bf16(v16bf a, v16bf b, v8f c) {
  return __builtin_amdgcn_wmma_f32_16x16x32_bf16(false, a, false, b, (short)0, c, false, false);
}
__device__ __forceinline__ unsigned short f2bf(float f) {
  unsigned int x = __builtin_bit_cast(unsigned int, f);
  x += 0x7fffu + ((x >> 16) & 1u);           // round-to-nearest-even
  return (unsigned short)(x >> 16);
}
__device__ __forceinline__ unsigned int pack2bf(float a, float b) {
  return (unsigned int)f2bf(a) | ((unsigned int)f2bf(b) << 16);
}

// ---------- constants ----------
// B=16 H=W=56 C=512 WS=7 SHIFT=3 NH=16 HD=32 N=49 NW=64 -> 1024 windows, 50176 tokens
#define SCALE_Q 0.17677669529663687f

// ---------- kernel 1: weight transpose+convert f32 -> bf16 ----------
__global__ void wconv_kernel(const float* __restrict__ Wqkv_a, const float* __restrict__ Wqkv_b,
                             const float* __restrict__ Wproj,
                             unsigned short* __restrict__ wqT, unsigned short* __restrict__ wkT,
                             unsigned short* __restrict__ wvT, unsigned short* __restrict__ wpT) {
  int id = blockIdx.x * blockDim.x + threadIdx.x;   // 0 .. 512*512-1
  int o = id >> 9, k = id & 511;
  wqT[o * 512 + k] = f2bf(Wqkv_b[k * 1536 + o]);            // q from rescaler stream
  wkT[o * 512 + k] = f2bf(Wqkv_a[k * 1536 + 512 + o]);      // k from rescaled stream
  wvT[o * 512 + k] = f2bf(Wqkv_a[k * 1536 + 1024 + o]);     // v from rescaled stream
  wpT[o * 512 + k] = f2bf(Wproj[k * 512 + o]);
}

// ---------- kernel 2: LayerNorm + roll(-3,-3) + window partition + bf16 ----------
__global__ void lnwin_kernel(const float* __restrict__ xa, const float* __restrict__ xb,
                             const float* __restrict__ ga, const float* __restrict__ ba,
                             const float* __restrict__ gb, const float* __restrict__ bb,
                             unsigned short* __restrict__ xw, unsigned short* __restrict__ yw) {
  int wave = (blockIdx.x * blockDim.x + threadIdx.x) >> 5;  // token id 0..50175
  int lane = threadIdx.x & 31;
  const float* src; const float* g; const float* bt; unsigned short* dst;
  if (blockIdx.y == 0) { src = xa; g = ga; bt = ba; dst = xw; }
  else                 { src = xb; g = gb; bt = bb; dst = yw; }
  int t = wave;
  int wh = t / 49, n = t % 49;
  int b = wh >> 6, wl = wh & 63;
  int wi = wl >> 3, wj = wl & 7;
  int ni = n / 7,  nj = n % 7;
  int h = (wi * 7 + ni + 3) % 56;      // roll(-SHIFT): rolled[p] = x[(p+SHIFT)%56]
  int w = (wj * 7 + nj + 3) % 56;
  const float* row = src + (size_t)(b * 3136 + h * 56 + w) * 512;
  int c0 = lane * 16;
  float x[16];
  float s = 0.f, s2 = 0.f;
#pragma unroll
  for (int i = 0; i < 16; ++i) { float v = row[c0 + i]; x[i] = v; s += v; s2 += v * v; }
#pragma unroll
  for (int d = 1; d < 32; d <<= 1) { s += __shfl_xor(s, d, 32); s2 += __shfl_xor(s2, d, 32); }
  float mean = s * (1.f / 512.f);
  float var  = s2 * (1.f / 512.f) - mean * mean;
  float rs   = rsqrtf(var + 1e-5f);
  unsigned int* drow = (unsigned int*)(dst + (size_t)t * 512 + c0);
#pragma unroll
  for (int j = 0; j < 8; ++j) {
    float y0 = (x[2 * j]     - mean) * rs * g[c0 + 2 * j]     + bt[c0 + 2 * j];
    float y1 = (x[2 * j + 1] - mean) * rs * g[c0 + 2 * j + 1] + bt[c0 + 2 * j + 1];
    drow[j] = pack2bf(y0, y1);
  }
}

// ---------- kernel 3: QKV GEMMs (one wave per (window, head)) ----------
// Rows 49..63 are clamped to row 48 (branchless); the garbage they produce is
// provably masked downstream: S^T rows m>=49 get -1e30 before softmax, S^T
// columns n>=49 are never stored, and v rows m>=49 are multiplied by P==0.
__global__ void __launch_bounds__(128) qkv_kernel(
    const unsigned short* __restrict__ xw, const unsigned short* __restrict__ yw,
    const unsigned short* __restrict__ wqT, const unsigned short* __restrict__ wkT,
    const unsigned short* __restrict__ wvT,
    const float* __restrict__ bqkv_a, const float* __restrict__ bqkv_b,
    unsigned short* __restrict__ qb, unsigned short* __restrict__ kb,
    unsigned short* __restrict__ vt) {
  int wave = (blockIdx.x * blockDim.x + threadIdx.x) >> 5;
  int lane = threadIdx.x & 31;
  int wh = wave >> 4, hH = wave & 15;
  int lanelo = lane & 15;
  int hi8  = (lane >> 4) * 8;
  int hi16 = (lane >> 4) * 16;
  size_t hb = (size_t)(wh * 16 + hH) * 2048;   // 64x32 (or 32x64) bf16 per head
  int outcol = hH * 32 + lanelo;

  // clamped token rows for the 4 M-tiles (branchless)
  size_t arow[4];
#pragma unroll
  for (int mt = 0; mt < 4; ++mt) {
    int rrow = mt * 16 + lanelo;
    rrow = (rrow < 49) ? rrow : 48;
    arow[mt] = (size_t)(wh * 49 + rrow) * 512 + hi8;
  }

  // ---- pass 1: q = yw @ WqT ----
  {
    v8f acc[4][2];
#pragma unroll
    for (int mt = 0; mt < 4; ++mt)
#pragma unroll
      for (int nt = 0; nt < 2; ++nt) acc[mt][nt] = vzero8();
    for (int ks = 0; ks < 16; ++ks) {
      int kbase = ks * 32;
      v16bf A[4];
#pragma unroll
      for (int mt = 0; mt < 4; ++mt) A[mt] = load_a_tile(yw + arow[mt] + kbase);
      v16bf Bv[2];
#pragma unroll
      for (int nt = 0; nt < 2; ++nt)
        Bv[nt] = load_b_tile(wqT + (size_t)(outcol + 16 * nt) * 512 + kbase + hi8);
#pragma unroll
      for (int mt = 0; mt < 4; ++mt)
#pragma unroll
        for (int nt = 0; nt < 2; ++nt)
          acc[mt][nt] = wmma_bf16(A[mt], Bv[nt], acc[mt][nt]);
    }
#pragma unroll
    for (int nt = 0; nt < 2; ++nt) {
      float bv = bqkv_b[outcol + 16 * nt];
      int d = 16 * nt + lanelo;
#pragma unroll
      for (int mt = 0; mt < 4; ++mt)
#pragma unroll
        for (int r = 0; r < 8; ++r) {
          int rowm = 16 * mt + hi8 + r;
          qb[hb + (size_t)rowm * 32 + d] = f2bf((acc[mt][nt][r] + bv) * SCALE_Q);
        }
    }
  }

  // ---- pass 2: k = xw @ WkT, v = xw @ WvT (shared A tiles) ----
  {
    v8f acck[4][2], accv[4][2];
#pragma unroll
    for (int mt = 0; mt < 4; ++mt)
#pragma unroll
      for (int nt = 0; nt < 2; ++nt) { acck[mt][nt] = vzero8(); accv[mt][nt] = vzero8(); }
    for (int ks = 0; ks < 16; ++ks) {
      int kbase = ks * 32;
      v16bf A[4];
#pragma unroll
      for (int mt = 0; mt < 4; ++mt) A[mt] = load_a_tile(xw + arow[mt] + kbase);
      v16bf Bk[2], Bv[2];
#pragma unroll
      for (int nt = 0; nt < 2; ++nt) {
        Bk[nt] = load_b_tile(wkT + (size_t)(outcol + 16 * nt) * 512 + kbase + hi8);
        Bv[nt] = load_b_tile(wvT + (size_t)(outcol + 16 * nt) * 512 + kbase + hi8);
      }
#pragma unroll
      for (int mt = 0; mt < 4; ++mt)
#pragma unroll
        for (int nt = 0; nt < 2; ++nt) {
          acck[mt][nt] = wmma_bf16(A[mt], Bk[nt], acck[mt][nt]);
          accv[mt][nt] = wmma_bf16(A[mt], Bv[nt], accv[mt][nt]);
        }
    }
#pragma unroll
    for (int nt = 0; nt < 2; ++nt) {
      float bvk = bqkv_a[512 + outcol + 16 * nt];
      float bvv = bqkv_a[1024 + outcol + 16 * nt];
      int d = 16 * nt + lanelo;
#pragma unroll
      for (int mt = 0; mt < 4; ++mt) {
        // k row-major [row(64)][d(32)]
#pragma unroll
        for (int r = 0; r < 8; ++r) {
          int rowm = 16 * mt + hi8 + r;
          kb[hb + (size_t)rowm * 32 + d] = f2bf(acck[mt][nt][r] + bvk);
        }
        // v transposed [d(32)][m(64)] -> pack pairs along m
#pragma unroll
        for (int j = 0; j < 4; ++j) {
          int rowm = 16 * mt + hi8 + 2 * j;
          *(unsigned int*)(vt + hb + (size_t)d * 64 + rowm) =
              pack2bf(accv[mt][nt][2 * j] + bvv, accv[mt][nt][2 * j + 1] + bvv);
        }
      }
    }
  }
}

// ---------- kernel 4: attention (one wave per (window, head)) ----------
__global__ void __launch_bounds__(128) attn_kernel(
    const unsigned short* __restrict__ qb, const unsigned short* __restrict__ kb,
    const unsigned short* __restrict__ vt, const float* __restrict__ rpb_table,
    unsigned short* __restrict__ ao) {
  __shared__ unsigned short Pl[4][64][72];   // P^T per wave slot, stride 72 halfs (144B, 16B-aligned)
  int wave  = (blockIdx.x * blockDim.x + threadIdx.x) >> 5;
  int wslot = threadIdx.x >> 5;
  int lane  = threadIdx.x & 31;
  int lanelo = lane & 15;
  int hi8  = (lane >> 4) * 8;
  int hi16 = (lane >> 4) * 16;
  int wh = wave >> 4, hH = wave & 15;
  size_t hb = (size_t)(wh * 16 + hH) * 2048;
  const unsigned short* q = qb + hb;
  const unsigned short* k = kb + hb;
  const unsigned short* v = vt + hb;

  // GEMM1: S^T[m,n] = sum_d k[m,d] * q[n,d]   (A = k rows, B = q rows)
  v16bf A[4], Bq[4];
#pragma unroll
  for (int mt = 0; mt < 4; ++mt)
    A[mt] = load_a_tile(k + (size_t)(16 * mt + lanelo) * 32 + hi8);
#pragma unroll
  for (int nt = 0; nt < 4; ++nt)
    Bq[nt] = load_b_tile(q + (size_t)(16 * nt + lanelo) * 32 + hi8);
  v8f S[4][4];
#pragma unroll
  for (int mt = 0; mt < 4; ++mt)
#pragma unroll
    for (int nt = 0; nt < 4; ++nt)
      S[mt][nt] = wmma_bf16(A[mt], Bq[nt], vzero8());

  // window geometry for shift mask
  int wl = wh & 63;
  int wi = wl >> 3, wj = wl & 7;

  // bias + mask + softmax over m (per column n of S^T), write P^T to LDS
#pragma unroll
  for (int nt = 0; nt < 4; ++nt) {
    int n = 16 * nt + lanelo;
    int ni = n / 7, nj = n % 7;
    int hn = wi * 7 + ni, wn = wj * 7 + nj;
    int rn = ((hn < 49) ? 0 : ((hn < 53) ? 1 : 2)) * 3 + ((wn < 49) ? 0 : ((wn < 53) ? 1 : 2));
    bool nvalid = (n < 49);
    float mx = -3.0e38f;
#pragma unroll
    for (int mt = 0; mt < 4; ++mt)
#pragma unroll
      for (int r = 0; r < 8; ++r) {
        int m = 16 * mt + hi8 + r;
        float val = S[mt][nt][r];
        if (m < 49) {
          if (nvalid) {
            int mi = m / 7, mj = m % 7;
            int hm = wi * 7 + mi, wm = wj * 7 + mj;
            int rm = ((hm < 49) ? 0 : ((hm < 53) ? 1 : 2)) * 3 + ((wm < 49) ? 0 : ((wm < 53) ? 1 : 2));
            int rel = (ni - mi + 6) * 13 + (nj - mj + 6);
            val += rpb_table[rel * 16 + hH];
            if (rm != rn) val += -100.0f;
          }
        } else {
          val = -1.0e30f;   // pad keys masked out
        }
        S[mt][nt][r] = val;
        mx = fmaxf(mx, val);
      }
    mx = fmaxf(mx, __shfl_xor(mx, 16, 32));
    float sum = 0.f;
#pragma unroll
    for (int mt = 0; mt < 4; ++mt)
#pragma unroll
      for (int r = 0; r < 8; ++r) {
        float p = __expf(S[mt][nt][r] - mx);
        S[mt][nt][r] = p;
        sum += p;
      }
    sum += __shfl_xor(sum, 16, 32);
    float inv = 1.0f / fmaxf(sum, 1e-20f);
#pragma unroll
    for (int mt = 0; mt < 4; ++mt)
#pragma unroll
      for (int j = 0; j < 4; ++j) {
        int m0 = 16 * mt + hi8 + 2 * j;
        *(unsigned int*)&Pl[wslot][n][m0] =
            pack2bf(S[mt][nt][2 * j] * inv, S[mt][nt][2 * j + 1] * inv);
      }
  }

  // GEMM2: O^T[d,n] = sum_m vT[d,m] * P^T[m,n]   (A = vT rows, B = P^T from LDS)
  v8f O[2][4];
#pragma unroll
  for (int mt = 0; mt < 2; ++mt)
#pragma unroll
    for (int nt = 0; nt < 4; ++nt) O[mt][nt] = vzero8();
#pragma unroll
  for (int kt = 0; kt < 2; ++kt) {
    v16bf Av[2];
#pragma unroll
    for (int mt = 0; mt < 2; ++mt)
      Av[mt] = load_a_tile(v + (size_t)(16 * mt + lanelo) * 64 + kt * 32 + hi8);
    v16bf Bp[4];
#pragma unroll
    for (int nt = 0; nt < 4; ++nt)
      Bp[nt] = load_b_tile(&Pl[wslot][16 * nt + lanelo][kt * 32 + hi8]);
#pragma unroll
    for (int mt = 0; mt < 2; ++mt)
#pragma unroll
      for (int nt = 0; nt < 4; ++nt)
        O[mt][nt] = wmma_bf16(Av[mt], Bp[nt], O[mt][nt]);
  }

  // store: ao[(wh*49+n)*512 + h*32 + d] (bf16), valid tokens only
#pragma unroll
  for (int nt = 0; nt < 4; ++nt) {
    int n = 16 * nt + lanelo;
    if (n < 49) {
      unsigned short* dst = ao + (size_t)(wh * 49 + n) * 512 + hH * 32;
#pragma unroll
      for (int mt = 0; mt < 2; ++mt)
#pragma unroll
        for (int j = 0; j < 4; ++j) {
          int d0 = 16 * mt + hi8 + 2 * j;
          *(unsigned int*)(dst + d0) = pack2bf(O[mt][nt][2 * j], O[mt][nt][2 * j + 1]);
        }
    }
  }
}

// ---------- kernel 5: projection GEMM + bias + window reverse + roll(+3) ----------
__global__ void __launch_bounds__(128) proj_kernel(
    const unsigned short* __restrict__ ao, const unsigned short* __restrict__ wpT,
    const float* __restrict__ bproj, float* __restrict__ outp) {
  int wave = (blockIdx.x * blockDim.x + threadIdx.x) >> 5;
  int lane = threadIdx.x & 31;
  int lanelo = lane & 15;
  int hi8  = (lane >> 4) * 8;
  int hi16 = (lane >> 4) * 16;
  int Mtile = wave >> 3;       // 0..3135 (50176/16)
  int g = wave & 7;            // group of 64 output cols
  size_t obase[8];
#pragma unroll
  for (int r = 0; r < 8; ++r) {
    int t = Mtile * 16 + hi8 + r;
    int wh = t / 49, n = t % 49;
    int b = wh >> 6, wl = wh & 63;
    int wi = wl >> 3, wj = wl & 7;
    int ni = n / 7,  nj = n % 7;
    int hh = (wi * 7 + ni + 3) % 56;   // window reverse + roll(+SHIFT)
    int ww = (wj * 7 + nj + 3) % 56;
    obase[r] = (size_t)(b * 3136 + hh * 56 + ww) * 512;
  }
  v8f acc[4];
#pragma unroll
  for (int nt = 0; nt < 4; ++nt) acc[nt] = vzero8();
  for (int ks = 0; ks < 16; ++ks) {
    const unsigned short* pa = ao + (size_t)(Mtile * 16 + lanelo) * 512 + ks * 32 + hi8;
    v16bf A = load_a_tile(pa);
#pragma unroll
    for (int nt = 0; nt < 4; ++nt) {
      int oc = g * 64 + 16 * nt + lanelo;
      v16bf Bv = load_b_tile(wpT + (size_t)oc * 512 + ks * 32 + hi8);
      acc[nt] = wmma_bf16(A, Bv, acc[nt]);
    }
  }
#pragma unroll
  for (int nt = 0; nt < 4; ++nt) {
    int oc = g * 64 + 16 * nt + lanelo;
    float bv = bproj[oc];
#pragma unroll
    for (int r = 0; r < 8; ++r)
      outp[obase[r] + oc] = acc[nt][r] + bv;
  }
}

// ---------- launch ----------
extern "C" void kernel_launch(void* const* d_in, const int* in_sizes, int n_in,
                              void* d_out, int out_size, void* d_ws, size_t ws_size,
                              hipStream_t stream) {
  (void)in_sizes; (void)n_in; (void)out_size; (void)ws_size;
  const float* rescaled  = (const float*)d_in[0];
  const float* rescaler  = (const float*)d_in[1];
  const float* gamma_a   = (const float*)d_in[2];
  const float* beta_a    = (const float*)d_in[3];
  const float* gamma_b   = (const float*)d_in[4];
  const float* beta_b    = (const float*)d_in[5];
  const float* Wqkv_a    = (const float*)d_in[6];
  const float* bqkv_a    = (const float*)d_in[7];
  const float* Wqkv_b    = (const float*)d_in[8];
  const float* bqkv_b    = (const float*)d_in[9];
  const float* rpb_table = (const float*)d_in[10];
  const float* Wproj     = (const float*)d_in[11];
  const float* bproj     = (const float*)d_in[12];
  float* outp = (float*)d_out;

  char* ws = (char*)d_ws;
  size_t off = 0;
  auto carve = [&](size_t bytes) { void* p = ws + off; off += (bytes + 255) & ~(size_t)255; return p; };
  unsigned short* xw  = (unsigned short*)carve(50176ull * 512 * 2);
  unsigned short* yw  = (unsigned short*)carve(50176ull * 512 * 2);
  unsigned short* qb  = (unsigned short*)carve(16384ull * 2048 * 2);
  unsigned short* kb  = (unsigned short*)carve(16384ull * 2048 * 2);
  unsigned short* vt  = (unsigned short*)carve(16384ull * 2048 * 2);
  unsigned short* wqT = (unsigned short*)carve(512ull * 512 * 2);
  unsigned short* wkT = (unsigned short*)carve(512ull * 512 * 2);
  unsigned short* wvT = (unsigned short*)carve(512ull * 512 * 2);
  unsigned short* wpT = (unsigned short*)carve(512ull * 512 * 2);
  unsigned short* ao  = xw;   // xw dead after qkv_kernel; reuse for attention output

  wconv_kernel<<<1024, 256, 0, stream>>>(Wqkv_a, Wqkv_b, Wproj, wqT, wkT, wvT, wpT);
  lnwin_kernel<<<dim3(6272, 2), 256, 0, stream>>>(rescaled, rescaler, gamma_a, beta_a,
                                                  gamma_b, beta_b, xw, yw);
  qkv_kernel<<<4096, 128, 0, stream>>>(xw, yw, wqT, wkT, wvT, bqkv_a, bqkv_b, qb, kb, vt);
  attn_kernel<<<4096, 128, 0, stream>>>(qb, kb, vt, rpb_table, ao);
  proj_kernel<<<6272, 128, 0, stream>>>(ao, wpT, bproj, outp);
}
